// AttentionBlock_46505905881687
// MI455X (gfx1250) — matrix operand order repaired and checked by reading, not compiled, not run on hardware
//
#include <hip/hip_runtime.h>

typedef __attribute__((ext_vector_type(16))) _Float16 v16h;
typedef __attribute__((ext_vector_type(8)))  float    v8f;
typedef int v4i __attribute__((vector_size(16)));     // gcc-style, matches builtin proto
typedef __attribute__((address_space(1))) v4i gv4i;   // global
typedef __attribute__((address_space(3))) v4i lv4i;   // LDS
typedef unsigned int u32;

#define BATCH 4
#define CH    128
#define NSP   4096
#define CPG   4
#define SCALE 0.08838834764831845f   // 1/sqrt(128)

union AFrag { u32 u[8]; v16h v; };

// A-matrix 16x32 f16 fragment (ISA 7.12.2): lane = M row, K pairs per VGPR.
__device__ __forceinline__ void load_af(const _Float16* base, int strideH, int lane, AFrag& f) {
  int r  = lane & 15;
  int kh = (lane >> 4) << 4;           // lanes 16-31 hold K=16..31
  const _Float16* p = base + r * strideH;
#pragma unroll
  for (int j = 0; j < 4; ++j) f.u[j]   = *(const u32*)(p + kh + 2*j);
#pragma unroll
  for (int j = 0; j < 4; ++j) f.u[4+j] = *(const u32*)(p + 8 + kh + 2*j);
}

// B-matrix 32x16 f16 fragment: lane = N col, kbase = (lane/16)*16, K pairs per VGPR.
// Source stored N-major with K contiguous: element (K, n) at base[n*strideH + K].
__device__ __forceinline__ void load_bf(const _Float16* base, int strideH, int lane, AFrag& f) {
  int n  = lane & 15;
  int kb = (lane >> 4) << 4;
  const _Float16* p = base + n * strideH + kb;
#pragma unroll
  for (int j = 0; j < 8; ++j) f.u[j] = *(const u32*)(p + 2*j);
}

// ---- CDNA5 async global->LDS (ASYNCcnt-tracked), 16 bytes per lane ----
__device__ __forceinline__ void async_copy16(_Float16* lds, const _Float16* g) {
#if __has_builtin(__builtin_amdgcn_global_load_async_to_lds_b128)
  __builtin_amdgcn_global_load_async_to_lds_b128(
      (gv4i*)(uintptr_t)g,
      (lv4i*)(u32)(uintptr_t)lds,
      0, 0);
#else
  u32 loff = (u32)(uintptr_t)lds;      // low 32 bits of generic LDS ptr = LDS offset
  asm volatile("global_load_async_to_lds_b128 %0, %1, off"
               :: "v"(loff), "v"(g) : "memory");
#endif
}

__device__ __forceinline__ void wait_async0() {
#if __has_builtin(__builtin_amdgcn_s_wait_asynccnt)
  __builtin_amdgcn_s_wait_asynccnt(0);
#else
  asm volatile("s_wait_asynccnt 0x0" ::: "memory");
#endif
}

// ---------------- weight cast ----------------
__global__ __launch_bounds__(256) void prep_kernel(const float* __restrict__ qkv_w,
    const float* __restrict__ out_w, _Float16* __restrict__ wq16, _Float16* __restrict__ wo16) {
  int i = blockIdx.x * 256 + threadIdx.x;
  if (i < 3*CH*CH) wq16[i] = (_Float16)qkv_w[i];
  int j = i - 3*CH*CH;
  if (j >= 0 && j < CH*CH) wo16[j] = (_Float16)out_w[j];
}

// ---------------- GroupNorm -> xnt[b][n][c] f16 ----------------
__global__ __launch_bounds__(256) void gn_kernel(const float* __restrict__ x,
    const float* __restrict__ gw, const float* __restrict__ gb, _Float16* __restrict__ xnt) {
  int bg = blockIdx.x, b = bg >> 5, g = bg & 31;
  const float* xp = x + (size_t)(b*CH + g*CPG) * NSP;
  int tid = threadIdx.x;
  float s = 0.f, ss = 0.f;
  for (int i = tid; i < CPG*NSP; i += 256) { float v = xp[i]; s += v; ss += v*v; }
#pragma unroll
  for (int off = 16; off > 0; off >>= 1) { s += __shfl_xor(s, off, 32); ss += __shfl_xor(ss, off, 32); }
  __shared__ float rs[8], rss[8], mv[2];
  int lane = tid & 31, wid = tid >> 5;
  if (lane == 0) { rs[wid] = s; rss[wid] = ss; }
  __syncthreads();
  if (tid == 0) {
    float S = 0.f, SS = 0.f;
    for (int i = 0; i < 8; ++i) { S += rs[i]; SS += rss[i]; }
    float mean = S / (float)(CPG*NSP);
    float var  = SS / (float)(CPG*NSP) - mean*mean;
    mv[0] = mean; mv[1] = rsqrtf(var + 1e-5f);
  }
  __syncthreads();
  float mean = mv[0], inv = mv[1];
  for (int i = tid; i < CPG*NSP; i += 256) {
    int cc = i >> 12, n = i & (NSP-1);
    int c = g*CPG + cc;
    float v = (xp[i] - mean) * inv * gw[c] + gb[c];
    xnt[((size_t)b*NSP + n)*CH + c] = (_Float16)v;
  }
}

// ---------------- QKV GEMM: (n x o) tiles, K = C = 128 ----------------
__global__ __launch_bounds__(256) void qkv_kernel(const _Float16* __restrict__ xnt,
    const _Float16* __restrict__ wq16, const float* __restrict__ qkv_b,
    _Float16* __restrict__ q16, _Float16* __restrict__ k16, _Float16* __restrict__ v16t) {
  int tid = threadIdx.x, lane = tid & 31, wid = tid >> 5;
  int ntile = blockIdx.x * 8 + wid;       // 0..1023 tiles over B*N rows
  int Rbase = ntile * 16;
  int obase = blockIdx.y * 16;            // 0..383 output channels
  v8f acc = {};
#pragma unroll
  for (int kk = 0; kk < 4; ++kk) {
    AFrag a, bf;
    load_af(xnt  + (size_t)Rbase*CH + kk*32, CH, lane, a);
    load_bf(wq16 + (size_t)obase*CH + kk*32, CH, lane, bf);  // B[c][o] = W[o][c]
    acc = __builtin_amdgcn_wmma_f32_16x16x32_f16(false, a.v, false, bf.v, (short)0, acc, false, false);
  }
  int o = obase + (lane & 15);
  float bias = qkv_b[o];
  int hi = (lane >> 4) * 8;
#pragma unroll
  for (int j = 0; j < 8; ++j) {
    int R = Rbase + j + hi;
    _Float16 h = (_Float16)(acc[j] + bias);
    if (o < CH)          q16[(size_t)R*CH + o] = h;
    else if (o < 2*CH)   k16[(size_t)R*CH + (o-CH)] = h;
    else {
      int b = R >> 12, n = R & (NSP-1);
      v16t[((size_t)b*CH + (o-2*CH))*NSP + n] = h;           // V transposed [c][n]
    }
  }
}

// ---------------- Flash attention: 4 waves x 16 query rows, async double-buffered KV ----------------
__global__ __launch_bounds__(128) void attn_kernel(const _Float16* __restrict__ q16,
    const _Float16* __restrict__ k16, const _Float16* __restrict__ v16t,
    _Float16* __restrict__ ao16) {
  __shared__ __attribute__((aligned(16))) _Float16 ldsK[2][32*CH];   // [buf][m][c]
  __shared__ __attribute__((aligned(16))) _Float16 ldsV[2][CH*32];   // [buf][c][m]
  __shared__ __attribute__((aligned(16))) _Float16 ldsP[4*16*32];    // per-wave P tile
  int tid = threadIdx.x, lane = tid & 31, wid = tid >> 5;
  int b  = blockIdx.x >> 6;
  int qb = (blockIdx.x & 63) * 64;
  int Rbase = b*NSP + qb + wid*16;

  const _Float16* kbase_g = k16  + (size_t)b*NSP*CH;
  const _Float16* vbase_g = v16t + (size_t)b*CH*NSP;

  // async-issue one 32-key chunk (K: 8KB, V: 8KB) into buffer `buf`
  auto issue_chunk = [&](int kc, int buf) {
#pragma unroll
    for (int i = 0; i < 4; ++i) {                 // K: 512 x 16B units, linear
      int u = tid + i*128;
      async_copy16(&ldsK[buf][u*8], kbase_g + (size_t)kc*CH + u*8);
    }
#pragma unroll
    for (int i = 0; i < 4; ++i) {                 // V: [c][m], 4 units per c-row
      int u = tid + i*128;
      int c = u >> 2, sub = u & 3;
      async_copy16(&ldsV[buf][u*8], vbase_g + (size_t)c*NSP + kc + sub*8);
    }
  };

  AFrag aq[4];
#pragma unroll
  for (int kk = 0; kk < 4; ++kk)
    load_af(q16 + (size_t)Rbase*CH + kk*32, CH, lane, aq[kk]);

  float m[8], l[8];
  v8f oacc[8];
#pragma unroll
  for (int j = 0; j < 8; ++j) { m[j] = -1e30f; l[j] = 0.f; }
#pragma unroll
  for (int ct = 0; ct < 8; ++ct) oacc[ct] = {};

  issue_chunk(0, 0);
  for (int it = 0; it < NSP/32; ++it) {
    int buf = it & 1;
    wait_async0();                 // chunk `it` resident in LDS (this wave's portion)
    __syncthreads();               // all waves' portions visible; prev compute done
    if (it + 1 < NSP/32) issue_chunk((it+1)*32, buf ^ 1);   // overlap DMA with compute

    // scores S = Q * K^T : two 16x16 subtiles over this 32-key chunk
    v8f s0 = {}, s1 = {};
#pragma unroll
    for (int kk = 0; kk < 4; ++kk) {
      AFrag bf; load_bf(ldsK[buf] + kk*32, CH, lane, bf);           // cols m=0..15
      s0 = __builtin_amdgcn_wmma_f32_16x16x32_f16(false, aq[kk].v, false, bf.v, (short)0, s0, false, false);
    }
#pragma unroll
    for (int kk = 0; kk < 4; ++kk) {
      AFrag bf; load_bf(ldsK[buf] + 16*CH + kk*32, CH, lane, bf);   // cols m=16..31
      s1 = __builtin_amdgcn_wmma_f32_16x16x32_f16(false, aq[kk].v, false, bf.v, (short)0, s1, false, false);
    }

    // online softmax: row = j + 8*(lane/16); row's 16 cols live in one 16-lane half
    float scj[8];
    int col = lane & 15, hi = lane >> 4;
    _Float16* pbase = ldsP + wid*512;
#pragma unroll
    for (int j = 0; j < 8; ++j) {
      float a0 = s0[j] * SCALE, a1 = s1[j] * SCALE;
      float t = fmaxf(a0, a1);
#pragma unroll
      for (int off = 8; off > 0; off >>= 1) t = fmaxf(t, __shfl_xor(t, off, 32));
      float mn = fmaxf(m[j], t);
      float sc = __expf(m[j] - mn);
      float p0 = __expf(a0 - mn), p1 = __expf(a1 - mn);
      float rs = p0 + p1;
#pragma unroll
      for (int off = 8; off > 0; off >>= 1) rs += __shfl_xor(rs, off, 32);
      m[j] = mn;
      l[j] = l[j]*sc + rs;
      scj[j] = sc;
      int row = j + hi*8;
      pbase[row*32 + col]      = (_Float16)p0;   // C-layout -> row-major P in LDS
      pbase[row*32 + 16 + col] = (_Float16)p1;
    }
#pragma unroll
    for (int ct = 0; ct < 8; ++ct)
#pragma unroll
      for (int j = 0; j < 8; ++j) oacc[ct][j] *= scj[j];

    // O += P * V   (P re-read from LDS in A-fragment layout)
    AFrag ap;
    load_af(pbase, 32, lane, ap);
#pragma unroll
    for (int ct = 0; ct < 8; ++ct) {
      AFrag bf; load_bf(ldsV[buf] + ct*16*32, 32, lane, bf);        // B[m][c], ldsV is [c][m]
      oacc[ct] = __builtin_amdgcn_wmma_f32_16x16x32_f16(false, ap.v, false, bf.v, (short)0, oacc[ct], false, false);
    }
    __syncthreads();               // all reads of buf done before it is re-filled at it+2
  }

  int col = lane & 15, hi = lane >> 4;
#pragma unroll
  for (int j = 0; j < 8; ++j) {
    float invl = 1.0f / l[j];
    int R = Rbase + j + hi*8;
#pragma unroll
    for (int ct = 0; ct < 8; ++ct)
      ao16[(size_t)R*CH + ct*16 + col] = (_Float16)(oacc[ct][j] * invl);
  }
}

// ---------------- Out projection + bias + residual ----------------
__global__ __launch_bounds__(256) void proj_kernel(const _Float16* __restrict__ ao16,
    const _Float16* __restrict__ wo16, const float* __restrict__ ob,
    const float* __restrict__ x, float* __restrict__ out) {
  int tid = threadIdx.x, lane = tid & 31, wid = tid >> 5;
  int ntile = blockIdx.x * 8 + wid;
  int Rbase = ntile * 16;
  int cbase = blockIdx.y * 16;
  v8f acc = {};
#pragma unroll
  for (int kk = 0; kk < 4; ++kk) {
    AFrag a, bf;
    load_af(ao16 + (size_t)Rbase*CH + kk*32, CH, lane, a);
    load_bf(wo16 + (size_t)cbase*CH + kk*32, CH, lane, bf);    // B[c'][c] = W[c][c']
    acc = __builtin_amdgcn_wmma_f32_16x16x32_f16(false, a.v, false, bf.v, (short)0, acc, false, false);
  }
  int c = cbase + (lane & 15);
  float bias = ob[c];
  int hi = (lane >> 4) * 8;
#pragma unroll
  for (int j = 0; j < 8; ++j) {
    int R = Rbase + j + hi;
    int bb = R >> 12, n = R & (NSP-1);
    size_t idx = ((size_t)bb*CH + c)*NSP + n;
    out[idx] = acc[j] + bias + x[idx];
  }
}

extern "C" void kernel_launch(void* const* d_in, const int* in_sizes, int n_in,
                              void* d_out, int out_size, void* d_ws, size_t ws_size,
                              hipStream_t stream) {
  (void)in_sizes; (void)n_in; (void)out_size; (void)ws_size;
  const float* x     = (const float*)d_in[0];
  const float* gn_w  = (const float*)d_in[1];
  const float* gn_b  = (const float*)d_in[2];
  const float* qkv_w = (const float*)d_in[3];
  const float* qkv_b = (const float*)d_in[4];
  const float* out_w = (const float*)d_in[5];
  const float* out_b = (const float*)d_in[6];
  float* out = (float*)d_out;

  char* ws = (char*)d_ws;
  const size_t HALF_PLANE = (size_t)BATCH * NSP * CH * 2;   // 4 MiB
  _Float16* wq16 = (_Float16*)(ws);
  _Float16* wo16 = (_Float16*)(ws + 98304);
  _Float16* xnt  = (_Float16*)(ws + 131072);
  _Float16* q16  = (_Float16*)(ws + 131072 + 1*HALF_PLANE);
  _Float16* k16  = (_Float16*)(ws + 131072 + 2*HALF_PLANE);
  _Float16* v16t = (_Float16*)(ws + 131072 + 3*HALF_PLANE);
  _Float16* ao16 = (_Float16*)(ws + 131072 + 4*HALF_PLANE);

  prep_kernel<<<256, 256, 0, stream>>>(qkv_w, out_w, wq16, wo16);
  gn_kernel<<<BATCH*32, 256, 0, stream>>>(x, gn_w, gn_b, xnt);
  qkv_kernel<<<dim3(128, 24), 256, 0, stream>>>(xnt, wq16, qkv_b, q16, k16, v16t);
  attn_kernel<<<BATCH*(NSP/64), 128, 0, stream>>>(q16, k16, v16t, ao16);
  proj_kernel<<<dim3(128, 8), 256, 0, stream>>>(ao16, wo16, out_b, x, out);
}
